// WindowAttention_37649683317324
// MI455X (gfx1250) — matrix operand order repaired and checked by reading, not compiled, not run on hardware
//
#include <hip/hip_runtime.h>
#include <hip/hip_bf16.h>

typedef __attribute__((ext_vector_type(16))) __bf16 v16bf;
typedef __attribute__((ext_vector_type(8)))  float  v8f;

#define DIM 384
#define HEADS 12
#define HEAD_DIM 32
#define NTOK 49          // tokens per window (7x7)
#define BW 4096          // batch windows
#define NW 64            // windows per image (mask count)
#define MROWS (BW * NTOK)        // 200704, GEMM M
#define MGROUPS (MROWS / 64)     // 3136 (exact)
#define QK_SCALE 0.17677669529663687f   // 32^-0.5

// ---------------------------------------------------------------------------
// Kernel 0: fp32 -> bf16 weight conversion (one-shot, tiny)
// ---------------------------------------------------------------------------
__global__ void cvt_f32_bf16_kernel(const float* __restrict__ src,
                                    __bf16* __restrict__ dst, int n) {
  int i = blockIdx.x * 256 + threadIdx.x;
  if (i < n) dst[i] = (__bf16)src[i];
}

// ---------------------------------------------------------------------------
// Kernel 1: gather relative-position bias  bias_pre[h][i][j]
// ---------------------------------------------------------------------------
__global__ void rpb_gather_kernel(const float* __restrict__ table,
                                  const int* __restrict__ relidx,
                                  float* __restrict__ bias_pre) {
  int idx = blockIdx.x * blockDim.x + threadIdx.x;   // over 49*49
  if (idx >= NTOK * NTOK) return;
  int rel = relidx[idx];
#pragma unroll
  for (int h = 0; h < HEADS; ++h)
    bias_pre[(size_t)h * NTOK * NTOK + idx] = table[rel * HEADS + h];
}

// ---------------------------------------------------------------------------
// Kernel 2: QKV GEMM  (200704 x 384) * (384 x 1152)^T + bias
// Block = 4 waves = 64 rows x 64 cols. X k-panels (64x32 fp32, 8KB) are
// double-buffered in LDS via GLOBAL_LOAD_ASYNC_TO_LDS_B128 (ASYNCcnt).
// Wave w owns output columns [ng*64 + w*16, +16); B fragments are single
// 32B bf16 vector loads from the pre-converted weight matrix.
// ---------------------------------------------------------------------------
__global__ __launch_bounds__(128) void gemm_qkv_kernel(
    const float* __restrict__ X, const __bf16* __restrict__ Wb,
    const float* __restrict__ bias,
    __bf16* __restrict__ qo, __bf16* __restrict__ ko, __bf16* __restrict__ vo) {
  __shared__ float xbuf[2][64][32];               // 2 x 8 KB

  const int mg   = blockIdx.x / 18;
  const int ng   = blockIdx.x % 18;
  const int wave = threadIdx.x >> 5;
  const int lane = threadIdx.x & 31;
  const int tid  = threadIdx.x;
  const int l16  = lane & 15;
  const bool hi  = (lane & 16) != 0;
  const size_t row0 = (size_t)mg * 64;

  // stage one 64x32 fp32 k-panel: 512 16B chunks, 4 per thread (4 per wave-issue)
  auto issue_panel = [&](int buf, int k0) {
#pragma unroll
    for (int j = 0; j < 4; ++j) {
      const int c      = j * 128 + tid;           // chunk id
      const int row    = c >> 3;                  // 8 chunks per row
      const int coloff = (c & 7) * 4;
      const unsigned voff =
          (unsigned)(((row0 + row) * DIM + (size_t)k0 + coloff) * 4u);
      const unsigned ldsa = (unsigned)(size_t)(&xbuf[buf][row][coloff]);
      asm volatile("global_load_async_to_lds_b128 %0, %1, %2"
                   :: "v"(ldsa), "v"(voff), "s"(X) : "memory");
    }
  };

  const int ncol = ng * 64 + wave * 16 + l16;     // B column (row of W)
  const __bf16* wrow = Wb + (size_t)ncol * DIM;

  v8f acc[4] = {v8f{}, v8f{}, v8f{}, v8f{}};
  issue_panel(0, 0);
  for (int ks = 0; ks < 12; ++ks) {
    const int k0 = ks * 32;
    if (ks + 1 < 12) {
      issue_panel((ks + 1) & 1, k0 + 32);
      asm volatile("s_wait_asynccnt 0x4" ::: "memory");  // panel ks complete
    } else {
      asm volatile("s_wait_asynccnt 0x0" ::: "memory");
    }
    __syncthreads();                              // all waves' panel visible
    const float(*xb)[32] = xbuf[ks & 1];
    const v16bf b = *(const v16bf*)(wrow + k0 + (hi ? 16 : 0));
    const int ka = hi ? 16 : 0;
#pragma unroll
    for (int ms = 0; ms < 4; ++ms) {
      const int arow = ms * 16 + l16;
      v16bf a;
#pragma unroll
      for (int i = 0; i < 8; ++i) {
        a[i]     = (__bf16)xb[arow][ka + i];
        a[8 + i] = (__bf16)xb[arow][ka + 8 + i];
      }
      acc[ms] = __builtin_amdgcn_wmma_f32_16x16x32_bf16(
          false, a, false, b, (short)0, acc[ms], false, false);
    }
    __syncthreads();                              // done reading, panel reusable
  }

  const float bval = bias[ncol];
  const int which = ncol / DIM;                   // 0=q 1=k 2=v
  const int head  = (ncol % DIM) >> 5;
  const int d     = ncol & 31;
  const float scl = (which == 0) ? QK_SCALE : 1.0f;
  __bf16* dst = (which == 0) ? qo : (which == 1) ? ko : vo;
#pragma unroll
  for (int ms = 0; ms < 4; ++ms)
#pragma unroll
    for (int r = 0; r < 8; ++r) {
      const size_t m = row0 + ms * 16 + r + (hi ? 8 : 0);
      const int bwin = (int)(m / NTOK);
      const int t    = (int)(m % NTOK);
      dst[(((size_t)bwin * HEADS + head) * NTOK + t) * HEAD_DIM + d] =
          (__bf16)((acc[ms][r] + bval) * scl);
    }
}

// ---------------------------------------------------------------------------
// Kernel 3: fused window attention, one block (4 waves) per (window, head).
// S = Q*K^T via 4 WMMAs (keys padded 49->64 with -inf), +bias +mask,
// softmax via shfl_xor over the 16-lane column group, P staged bf16 in LDS,
// O = P*V via 2x2 WMMAs, 1/l normalization register-local.
// ---------------------------------------------------------------------------
__global__ __launch_bounds__(128) void attn_kernel(
    const __bf16* __restrict__ Q, const __bf16* __restrict__ K,
    const __bf16* __restrict__ V, const float* __restrict__ bias_pre,
    const float* __restrict__ mask, __bf16* __restrict__ O) {
  __shared__ __bf16 pTile[4][16][64];             // per-wave P tiles, 8 KB

  const int bh = blockIdx.x;
  const int b  = bh / HEADS;
  const int h  = bh % HEADS;
  const int w  = b & (NW - 1);
  const int qt   = threadIdx.x >> 5;
  const int lane = threadIdx.x & 31;
  const int l16  = lane & 15;
  const bool hiL = (lane & 16) != 0;
  const int g8   = hiL ? 8 : 0;

  const __bf16* q = Q + (size_t)bh * NTOK * HEAD_DIM;
  const __bf16* k = K + (size_t)bh * NTOK * HEAD_DIM;
  const __bf16* v = V + (size_t)bh * NTOK * HEAD_DIM;

  // ---- Q fragment (A-matrix, 16x32 over head_dim) ----
  v16bf aq;
  {
    const int t  = qt * 16 + l16;
    const bool tv = t < NTOK;
    const int ka = hiL ? 16 : 0;
#pragma unroll
    for (int i = 0; i < 8; ++i) {
      aq[i]     = tv ? q[t * HEAD_DIM + ka + i]      : (__bf16)0.0f;
      aq[8 + i] = tv ? q[t * HEAD_DIM + ka + 8 + i] : (__bf16)0.0f;
    }
  }

  // ---- S = Q * K^T  (4 key tiles) ----
  v8f s[4];
#pragma unroll
  for (int kt = 0; kt < 4; ++kt) {
    const int key = kt * 16 + l16;
    const bool kv = key < NTOK;
    const int db  = hiL ? 16 : 0;
    v16bf bk;
#pragma unroll
    for (int i = 0; i < 16; ++i)
      bk[i] = kv ? k[key * HEAD_DIM + db + i] : (__bf16)0.0f;
    v8f z = {};
    s[kt] = __builtin_amdgcn_wmma_f32_16x16x32_bf16(
        false, aq, false, bk, (short)0, z, false, false);
  }

  // ---- bias + mask + softmax (row = r + g8 within tile) ----
  float lsum[8];
#pragma unroll
  for (int r = 0; r < 8; ++r) {
    const int rit = r + g8;
    const int t   = qt * 16 + rit;
    const bool tv = t < NTOK;
    float sv[4];
    float mx = -3.0e38f;
#pragma unroll
    for (int kt = 0; kt < 4; ++kt) {
      const int col = kt * 16 + l16;
      float x = -3.0e38f;
      if (tv && col < NTOK) {
        x = s[kt][r] + bias_pre[((size_t)h * NTOK + t) * NTOK + col]
                     + mask[((size_t)w * NTOK + t) * NTOK + col];
      }
      sv[kt] = x;
      mx = fmaxf(mx, x);
    }
#pragma unroll
    for (int off = 8; off >= 1; off >>= 1)
      mx = fmaxf(mx, __shfl_xor(mx, off, 32));
    float sum = 0.0f;
#pragma unroll
    for (int kt = 0; kt < 4; ++kt) {
      const int col = kt * 16 + l16;
      float p = 0.0f;
      if (sv[kt] > -1.0e38f) p = __expf(sv[kt] - mx);
      pTile[qt][rit][col] = (__bf16)p;
      sum += p;
    }
#pragma unroll
    for (int off = 8; off >= 1; off >>= 1)
      sum += __shfl_xor(sum, off, 32);
    lsum[r] = sum;
  }
  __syncthreads();

  // ---- O = P * V  (2 d-tiles x 2 k-steps) ----
#pragma unroll
  for (int dt = 0; dt < 2; ++dt) {
    v8f o = {};
#pragma unroll
    for (int ks = 0; ks < 2; ++ks) {
      const int k0 = ks * 32;
      v16bf ap, bv;
      const int ka = k0 + (hiL ? 16 : 0);
#pragma unroll
      for (int i = 0; i < 8; ++i) {
        ap[i]     = pTile[qt][l16][ka + i];
        ap[8 + i] = pTile[qt][l16][ka + 8 + i];
      }
      const int d     = dt * 16 + l16;
      const int kbase = k0 + (hiL ? 16 : 0);
#pragma unroll
      for (int i = 0; i < 16; ++i) {
        const int key = kbase + i;
        bv[i] = (key < NTOK) ? v[key * HEAD_DIM + d] : (__bf16)0.0f;
      }
      o = __builtin_amdgcn_wmma_f32_16x16x32_bf16(
          false, ap, false, bv, (short)0, o, false, false);
    }
#pragma unroll
    for (int r = 0; r < 8; ++r) {
      const int rit = r + g8;
      const int t   = qt * 16 + rit;
      if (t < NTOK) {
        const float inv = (lsum[r] > 0.0f) ? 1.0f / lsum[r] : 0.0f;
        const int ch = h * HEAD_DIM + dt * 16 + l16;
        O[((size_t)b * NTOK + t) * DIM + ch] = (__bf16)(o[r] * inv);
      }
    }
  }
}

// ---------------------------------------------------------------------------
// Kernel 4: proj GEMM  (200704 x 384) * (384 x 384)^T + bias -> fp32 d_out
// Same async double-buffered structure; A panel is bf16 (64x32 = 4KB).
// ---------------------------------------------------------------------------
__global__ __launch_bounds__(128) void gemm_proj_kernel(
    const __bf16* __restrict__ A, const __bf16* __restrict__ Wb,
    const float* __restrict__ bias, float* __restrict__ out) {
  __shared__ __bf16 abuf[2][64][32];              // 2 x 4 KB

  const int mg   = blockIdx.x / 6;
  const int ng   = blockIdx.x % 6;
  const int wave = threadIdx.x >> 5;
  const int lane = threadIdx.x & 31;
  const int tid  = threadIdx.x;
  const int l16  = lane & 15;
  const bool hi  = (lane & 16) != 0;
  const size_t row0 = (size_t)mg * 64;

  // stage one 64x32 bf16 k-panel: 256 16B chunks, 2 per thread
  auto issue_panel = [&](int buf, int k0) {
#pragma unroll
    for (int j = 0; j < 2; ++j) {
      const int c      = j * 128 + tid;           // chunk id
      const int row    = c >> 2;                  // 4 chunks per row
      const int coloff = (c & 3) * 8;
      const unsigned voff =
          (unsigned)(((row0 + row) * DIM + (size_t)k0 + coloff) * 2u);
      const unsigned ldsa = (unsigned)(size_t)(&abuf[buf][row][coloff]);
      asm volatile("global_load_async_to_lds_b128 %0, %1, %2"
                   :: "v"(ldsa), "v"(voff), "s"(A) : "memory");
    }
  };

  const int ncol = ng * 64 + wave * 16 + l16;
  const __bf16* wrow = Wb + (size_t)ncol * DIM;

  v8f acc[4] = {v8f{}, v8f{}, v8f{}, v8f{}};
  issue_panel(0, 0);
  for (int ks = 0; ks < 12; ++ks) {
    const int k0 = ks * 32;
    if (ks + 1 < 12) {
      issue_panel((ks + 1) & 1, k0 + 32);
      asm volatile("s_wait_asynccnt 0x2" ::: "memory");
    } else {
      asm volatile("s_wait_asynccnt 0x0" ::: "memory");
    }
    __syncthreads();
    const __bf16(*ab)[32] = abuf[ks & 1];
    const v16bf b = *(const v16bf*)(wrow + k0 + (hi ? 16 : 0));
    const int ka = hi ? 16 : 0;
#pragma unroll
    for (int ms = 0; ms < 4; ++ms) {
      const int arow = ms * 16 + l16;
      v16bf a;
#pragma unroll
      for (int i = 0; i < 8; ++i) {
        a[i]     = ab[arow][ka + i];
        a[8 + i] = ab[arow][ka + 8 + i];
      }
      acc[ms] = __builtin_amdgcn_wmma_f32_16x16x32_bf16(
          false, a, false, b, (short)0, acc[ms], false, false);
    }
    __syncthreads();
  }

  const float bval = bias[ncol];
#pragma unroll
  for (int ms = 0; ms < 4; ++ms)
#pragma unroll
    for (int r = 0; r < 8; ++r) {
      const size_t m = row0 + ms * 16 + r + (hi ? 8 : 0);
      out[m * DIM + ncol] = acc[ms][r] + bval;
    }
}

// ---------------------------------------------------------------------------
// Host-side launcher
// ---------------------------------------------------------------------------
extern "C" void kernel_launch(void* const* d_in, const int* in_sizes, int n_in,
                              void* d_out, int out_size, void* d_ws, size_t ws_size,
                              hipStream_t stream) {
  const float* x      = (const float*)d_in[0];
  const float* amask  = (const float*)d_in[1];
  const float* qkv_w  = (const float*)d_in[2];
  const float* qkv_b  = (const float*)d_in[3];
  const float* proj_w = (const float*)d_in[4];
  const float* proj_b = (const float*)d_in[5];
  const float* rpb    = (const float*)d_in[6];
  const int*   relidx = (const int*)d_in[7];
  float* out = (float*)d_out;

  // Workspace layout
  const size_t PER_T = (size_t)BW * HEADS * NTOK * HEAD_DIM;  // 77,070,336
  char* p = (char*)d_ws;
  __bf16* q_ws  = (__bf16*)p;  p += PER_T * 2;
  __bf16* k_ws  = (__bf16*)p;  p += PER_T * 2;
  __bf16* v_ws  = (__bf16*)p;  p += PER_T * 2;
  __bf16* ao_ws = (__bf16*)p;  p += (size_t)MROWS * DIM * 2;  // (200704,384) bf16
  float*  bias_pre = (float*)p;  p += (size_t)HEADS * NTOK * NTOK * 4;
  __bf16* wqkv_bf  = (__bf16*)p; p += (size_t)3 * DIM * DIM * 2;
  __bf16* wproj_bf = (__bf16*)p;

  // 0) weights -> bf16 (one-shot)
  cvt_f32_bf16_kernel<<<(3 * DIM * DIM + 255) / 256, 256, 0, stream>>>(
      qkv_w, wqkv_bf, 3 * DIM * DIM);
  cvt_f32_bf16_kernel<<<(DIM * DIM + 255) / 256, 256, 0, stream>>>(
      proj_w, wproj_bf, DIM * DIM);

  // 1) relative-position bias gather
  rpb_gather_kernel<<<(NTOK * NTOK + 127) / 128, 128, 0, stream>>>(
      rpb, relidx, bias_pre);

  // 2) QKV GEMM: 3136 row-groups x 18 col-groups
  gemm_qkv_kernel<<<MGROUPS * 18, 128, 0, stream>>>(
      x, wqkv_bf, qkv_b, q_ws, k_ws, v_ws);

  // 3) fused attention: one block per (window, head)
  attn_kernel<<<BW * HEADS, 128, 0, stream>>>(
      q_ws, k_ws, v_ws, bias_pre, amask, ao_ws);

  // 4) proj GEMM: 3136 row-groups x 6 col-groups
  gemm_proj_kernel<<<MGROUPS * 6, 128, 0, stream>>>(
      ao_ws, wproj_bf, proj_b, out);
}
